// CausalSelfAttention_1142461301350
// MI455X (gfx1250) — compile-verified, matrix operand order checked by reading, not run on hardware
//
#include <hip/hip_runtime.h>

#define BB 4
#define SS 2048
#define DD 1024
#define HH 16
#define DH 64
#define D3 3072

typedef __attribute__((ext_vector_type(16))) __bf16 v16bf;
typedef __attribute__((ext_vector_type(8)))  __bf16 v8bf;
typedef __attribute__((ext_vector_type(8)))  float  v8f;

union Frag16 { v16bf v; v8bf h[2]; };

// Load a 16x32 bf16 A-style fragment (row-major source, leading dim ld halfs).
// Per CDNA5 ISA 16-bit A layout: lane = (m | grp<<4); lane holds K chunks
// [8*grp, 8*grp+8) in v0..3 and [16+8*grp, 16+8*grp+8) in v4..7 (packed pairs).
__device__ __forceinline__ v16bf load_frag(const __bf16* base, int ld, int lane) {
    int m   = lane & 15;
    int grp = lane >> 4;
    const __bf16* p = base + (size_t)m * ld + 8 * grp;
    Frag16 f;
    f.h[0] = *(const v8bf*)(p);
    f.h[1] = *(const v8bf*)(p + 16);
    return f.v;
}

__global__ void f32_to_bf16_kernel(const float* __restrict__ src,
                                   __bf16* __restrict__ dst, int n) {
    int i = blockIdx.x * blockDim.x + threadIdx.x;
    if (i < n) dst[i] = (__bf16)src[i];
}

// ---------------------------------------------------------------------------
// QKV GEMM: qkv[B*S, 3D] = xb[B*S, D] @ wa[3D, D]^T
// One wave per 64x32 macro-tile (4x2 subtiles): 8 WMMAs per 6 fragment loads.
// Tile coordinates are forced into SGPRs via readfirstlane so the q/k/v
// dispatch lowers to scalar branches (once per 16-col subtile, not per store).
// q,k -> [B*H, S, Dh]; v -> TRANSPOSED [B*H, Dh, S] for vectorized P*V loads.
// ---------------------------------------------------------------------------
__global__ void __launch_bounds__(256)
qkv_gemm_kernel(const __bf16* __restrict__ xb,
                const __bf16* __restrict__ wa,
                __bf16* __restrict__ qb,
                __bf16* __restrict__ kb,
                __bf16* __restrict__ vtb) {
    const int lane = threadIdx.x & 31;
    const int wave = threadIdx.x >> 5;
    const int tile = __builtin_amdgcn_readfirstlane(
        blockIdx.x * (blockDim.x >> 5) + wave);          // wave-uniform -> SGPR
    const int colTiles = D3 / 32;            // 96
    const int rowTile = tile / colTiles;
    const int colTile = tile % colTiles;
    const int row0 = rowTile * 64;           // index into B*S (scalar)
    const int col0 = colTile * 32;           // index into 3D   (scalar)

    v8f acc[4][2] = {};
    const __bf16* aBase = xb + (size_t)row0 * DD;
    const __bf16* bBase = wa + (size_t)col0 * DD;   // wa is [out,in] == B^T layout

    for (int k0 = 0; k0 < DD; k0 += 32) {
        if (k0 + 32 < DD) {
            __builtin_prefetch(aBase + k0 + 32, 0, 1);
            __builtin_prefetch(bBase + k0 + 32, 0, 1);
        }
        v16bf a[4];
        #pragma unroll
        for (int i = 0; i < 4; ++i)
            a[i] = load_frag(aBase + (size_t)(16 * i) * DD + k0, DD, lane);
        #pragma unroll
        for (int j = 0; j < 2; ++j) {
            v16bf b = load_frag(bBase + (size_t)(16 * j) * DD + k0, DD, lane);
            #pragma unroll
            for (int i = 0; i < 4; ++i)
                acc[i][j] = __builtin_amdgcn_wmma_f32_16x16x32_bf16(
                    false, a[i], false, b, (short)0, acc[i][j], false, false);
        }
    }

    // ---- epilogue: scalar dispatch per subtile, plain strided stores ----
    const int nloc = lane & 15;
    const int grp  = lane >> 4;
    const int bb = row0 >> 11;               // batch  (scalar; 64 | 2048)
    const int s0 = row0 & (SS - 1);          // seq    (scalar)
    #pragma unroll
    for (int j = 0; j < 2; ++j) {
        const int colj  = col0 + 16 * j;     // scalar
        const int which = colj >> 10;        // scalar: 0=q 1=k 2=v
        const int d0    = colj & (DD - 1);   // scalar
        const int h     = d0 >> 6;           // scalar (16 | 64 -> const in subtile)
        const int dh    = (d0 & (DH - 1)) + nloc;    // per-lane
        if (which == 2) {
            // transposed: [B*H, Dh, S]
            __bf16* base = vtb + ((size_t)(bb * HH + h) * DH + dh) * SS + s0;
            #pragma unroll
            for (int i = 0; i < 4; ++i)
                #pragma unroll
                for (int r = 0; r < 8; ++r)
                    base[16 * i + r + 8 * grp] = (__bf16)acc[i][j][r];
        } else {
            __bf16* dst  = (which == 0) ? qb : kb;
            __bf16* base = dst + ((size_t)(bb * HH + h) * SS + s0) * DH + dh;
            #pragma unroll
            for (int i = 0; i < 4; ++i)
                #pragma unroll
                for (int r = 0; r < 8; ++r)
                    base[(size_t)(16 * i + r + 8 * grp) * DH] = (__bf16)acc[i][j][r];
        }
    }
}

// ---------------------------------------------------------------------------
// Flash attention: one wave handles (b,h) x 16-query tile, streams 32-key
// tiles with online softmax.  Q/K bf16 [B*H, S, Dh]; V bf16 [B*H, Dh, S].
// ---------------------------------------------------------------------------
__global__ void attn_kernel(const __bf16* __restrict__ qb,
                            const __bf16* __restrict__ kb,
                            const __bf16* __restrict__ vtb,
                            __bf16* __restrict__ yb) {
    constexpr int WPB = 8;
    __shared__ __align__(16) float  scLds[WPB][16 * 32];
    __shared__ __align__(16) __bf16 pLds [WPB][16 * 32];
    __shared__ float rowMax[WPB][16];
    __shared__ float rowSum[WPB][16];
    __shared__ float alphaL[WPB][16];

    const int lane = threadIdx.x & 31;
    const int wave = threadIdx.x >> 5;
    const int wid  = __builtin_amdgcn_readfirstlane(blockIdx.x * WPB + wave);
    const int qt   = wid & (SS / 16 - 1);
    const int bh   = wid / (SS / 16);
    const int q0   = qt * 16;

    const __bf16* qPtr = qb  + (size_t)bh * SS * DH;
    const __bf16* kPtr = kb  + (size_t)bh * SS * DH;
    const __bf16* vPtr = vtb + (size_t)bh * DH * SS;   // [Dh, S]

    const v16bf aQ0 = load_frag(qPtr + (size_t)q0 * DH + 0,  DH, lane);
    const v16bf aQ1 = load_frag(qPtr + (size_t)q0 * DH + 32, DH, lane);

    v8f O[4] = {{}, {}, {}, {}};
    if (lane < 16) {
        rowMax[wave][lane] = -INFINITY;
        rowSum[wave][lane] = 0.0f;
    }
    const int nloc = lane & 15;
    const int grp  = lane >> 4;

    const int ktMax = (q0 + 15) / 32;
    for (int kt = 0; kt <= ktMax; ++kt) {
        const int k0 = kt * 32;
        // --- scores: S[16 x 32] = scale * Q Kt, causal masked -----------
        #pragma unroll
        for (int sub = 0; sub < 2; ++sub) {
            v16bf b0 = load_frag(kPtr + (size_t)(k0 + sub * 16) * DH + 0,  DH, lane);
            v16bf b1 = load_frag(kPtr + (size_t)(k0 + sub * 16) * DH + 32, DH, lane);
            v8f c = {};
            c = __builtin_amdgcn_wmma_f32_16x16x32_bf16(
                false, aQ1, false, b1, (short)0, c, false, false);
            c = __builtin_amdgcn_wmma_f32_16x16x32_bf16(
                false, aQ0, false, b0, (short)0, c, false, false);
            #pragma unroll
            for (int r = 0; r < 8; ++r) {
                int m   = r + 8 * grp;
                int key = k0 + sub * 16 + nloc;
                float s = c[r] * 0.125f;                 // 1/sqrt(64)
                if (key > q0 + m) s = -INFINITY;
                scLds[wave][m * 32 + sub * 16 + nloc] = s;
            }
        }
        asm volatile("s_wait_dscnt 0x0" ::: "memory");
        __builtin_amdgcn_wave_barrier();

        // --- online softmax (lanes 0..15, one row each) -----------------
        if (lane < 16) {
            int m = lane;
            float mx = rowMax[wave][m];
            float tmax = mx;
            #pragma unroll
            for (int n = 0; n < 32; ++n)
                tmax = fmaxf(tmax, scLds[wave][m * 32 + n]);
            float alpha = __expf(mx - tmax);
            float sum = 0.0f;
            #pragma unroll
            for (int n = 0; n < 32; ++n) {
                float p = __expf(scLds[wave][m * 32 + n] - tmax);
                pLds[wave][m * 32 + n] = (__bf16)p;
                sum += p;
            }
            rowSum[wave][m] = rowSum[wave][m] * alpha + sum;
            rowMax[wave][m] = tmax;
            alphaL[wave][m] = alpha;
        }
        asm volatile("s_wait_dscnt 0x0" ::: "memory");
        __builtin_amdgcn_wave_barrier();

        // --- rescale running O by alpha(row) ----------------------------
        #pragma unroll
        for (int r = 0; r < 8; ++r) {
            float al = alphaL[wave][r + 8 * grp];
            O[0][r] *= al; O[1][r] *= al; O[2][r] *= al; O[3][r] *= al;
        }

        // --- P fragment (16x32) from LDS --------------------------------
        Frag16 pf;
        {
            const __bf16* p = &pLds[wave][nloc * 32 + 8 * grp];
            pf.h[0] = *(const v8bf*)(p);
            pf.h[1] = *(const v8bf*)(p + 16);
        }
        // --- O[16x64] += P[16x32] x V[32x64]; V^T rows are contiguous ---
        #pragma unroll
        for (int j = 0; j < 4; ++j) {
            v16bf vf = load_frag(vPtr + (size_t)(j * 16) * SS + k0, SS, lane);
            O[j] = __builtin_amdgcn_wmma_f32_16x16x32_bf16(
                false, pf.v, false, vf, (short)0, O[j], false, false);
        }
        asm volatile("s_wait_dscnt 0x0" ::: "memory");
        __builtin_amdgcn_wave_barrier();
    }

    // --- normalize and store context [B, S, D] bf16 ---------------------
    const int b = bh / HH;
    const int h = bh % HH;
    #pragma unroll
    for (int r = 0; r < 8; ++r) {
        int m = r + 8 * grp;
        float inv = 1.0f / rowSum[wave][m];
        size_t base = ((size_t)b * SS + (q0 + m)) * DD + h * DH;
        #pragma unroll
        for (int j = 0; j < 4; ++j)
            yb[base + j * 16 + nloc] = (__bf16)(O[j][r] * inv);
    }
}

// ---------------------------------------------------------------------------
// Output projection: out[B*S, D] = yb[B*S, D] @ wp[D, D]^T, fp32 store.
// One wave per 64x32 macro-tile (4x2 register blocking).
// ---------------------------------------------------------------------------
__global__ void __launch_bounds__(256)
proj_gemm_kernel(const __bf16* __restrict__ yb,
                 const __bf16* __restrict__ wp,
                 float* __restrict__ out) {
    const int lane = threadIdx.x & 31;
    const int wave = threadIdx.x >> 5;
    const int tile = __builtin_amdgcn_readfirstlane(
        blockIdx.x * (blockDim.x >> 5) + wave);
    const int colTiles = DD / 32;            // 32
    const int rowTile = tile / colTiles;
    const int colTile = tile % colTiles;
    const int row0 = rowTile * 64;
    const int col0 = colTile * 32;

    v8f acc[4][2] = {};
    const __bf16* aBase = yb + (size_t)row0 * DD;
    const __bf16* bBase = wp + (size_t)col0 * DD;

    for (int k0 = 0; k0 < DD; k0 += 32) {
        if (k0 + 32 < DD) {
            __builtin_prefetch(aBase + k0 + 32, 0, 1);
            __builtin_prefetch(bBase + k0 + 32, 0, 1);
        }
        v16bf a[4];
        #pragma unroll
        for (int i = 0; i < 4; ++i)
            a[i] = load_frag(aBase + (size_t)(16 * i) * DD + k0, DD, lane);
        #pragma unroll
        for (int j = 0; j < 2; ++j) {
            v16bf b = load_frag(bBase + (size_t)(16 * j) * DD + k0, DD, lane);
            #pragma unroll
            for (int i = 0; i < 4; ++i)
                acc[i][j] = __builtin_amdgcn_wmma_f32_16x16x32_bf16(
                    false, a[i], false, b, (short)0, acc[i][j], false, false);
        }
    }

    const int nloc = lane & 15;
    const int grp  = lane >> 4;
    float* base = out + (size_t)row0 * DD + col0 + nloc;
    #pragma unroll
    for (int i = 0; i < 4; ++i)
        #pragma unroll
        for (int j = 0; j < 2; ++j)
            #pragma unroll
            for (int r = 0; r < 8; ++r)
                base[(size_t)(16 * i + r + 8 * grp) * DD + 16 * j] = acc[i][j][r];
}

// ---------------------------------------------------------------------------
extern "C" void kernel_launch(void* const* d_in, const int* in_sizes, int n_in,
                              void* d_out, int out_size, void* d_ws, size_t ws_size,
                              hipStream_t stream) {
    const float* x      = (const float*)d_in[0];   // [4, 2048, 1024]
    const float* w_attn = (const float*)d_in[1];   // [3072, 1024]
    const float* w_proj = (const float*)d_in[2];   // [1024, 1024]
    float* out = (float*)d_out;                    // [4, 2048, 1024]

    const size_t nX   = (size_t)BB * SS * DD;      // 8388608
    const size_t nWA  = (size_t)D3 * DD;           // 3145728
    const size_t nWP  = (size_t)DD * DD;           // 1048576
    const size_t nQKV = nX;                        // per q/k/v buffer

    __bf16* xb  = (__bf16*)d_ws;
    __bf16* wa  = xb  + nX;
    __bf16* wp  = wa  + nWA;
    __bf16* qb  = wp  + nWP;
    __bf16* kb  = qb  + nQKV;
    __bf16* vtb = kb  + nQKV;      // [B*H, Dh, S] (transposed)
    __bf16* ybf = vtb + nQKV;

    // 1) fp32 -> bf16 converts
    f32_to_bf16_kernel<<<(int)((nX  + 255) / 256), 256, 0, stream>>>(x,      xb, (int)nX);
    f32_to_bf16_kernel<<<(int)((nWA + 255) / 256), 256, 0, stream>>>(w_attn, wa, (int)nWA);
    f32_to_bf16_kernel<<<(int)((nWP + 255) / 256), 256, 0, stream>>>(w_proj, wp, (int)nWP);

    // 2) QKV GEMM: (8192/64)*(3072/32) = 12288 macro-tiles / 8 waves per block
    qkv_gemm_kernel<<<1536, 256, 0, stream>>>(xb, wa, qb, kb, vtb);

    // 3) attention: 4*16*128 = 8192 waves / 8 per block
    attn_kernel<<<1024, 256, 0, stream>>>(qb, kb, vtb, ybf);

    // 4) projection: (8192/64)*(1024/32) = 4096 macro-tiles / 8 per block
    proj_gemm_kernel<<<512, 256, 0, stream>>>(ybf, wp, out);
}
